// CAM_85770496901546
// MI455X (gfx1250) — matrix operand order repaired and checked by reading, not compiled, not run on hardware
//
#include <hip/hip_runtime.h>

// CDNA5 / gfx1250: wave32, WMMA 16x16x32 bf16 with f32 accumulate.
typedef __attribute__((ext_vector_type(16))) __bf16 v16bf;
typedef __attribute__((ext_vector_type(8)))  float  v8f;

union B16x16 { v16bf v; uint4 q[2]; };

// Async global->LDS (ASYNCcnt) path, device pass only, with sync fallback.
#if defined(__AMDGCN__) && defined(__has_builtin)
#  if __has_builtin(__builtin_amdgcn_global_load_async_to_lds_b128)
#    define HAVE_ASYNC_LDS 1
#  endif
#endif
#ifndef HAVE_ASYNC_LDS
#  define HAVE_ASYNC_LDS 0
#endif

#if HAVE_ASYNC_LDS
// Builtin prototype (from hipcc diagnostic): (AS1 v4i*, AS3 v4i*, Imm, Imm)
typedef int v4i __attribute__((vector_size(16)));
typedef __attribute__((address_space(1))) v4i* gv4i_p;   // global int4*
typedef __attribute__((address_space(3))) v4i* lv4i_p;   // LDS int4*
#endif

__device__ __forceinline__ void wait_async0() {
#if HAVE_ASYNC_LDS
#  if __has_builtin(__builtin_amdgcn_s_wait_asynccnt)
  __builtin_amdgcn_s_wait_asynccnt(0);
#  else
  asm volatile("s_wait_asynccnt 0x0" ::: "memory");
#  endif
#endif
}

__device__ __forceinline__ __bf16 f2bf(float f) {
  unsigned u = __float_as_uint(f);
  unsigned r = u + 0x7FFFu + ((u >> 16) & 1u);   // round-to-nearest-even
  unsigned short h = (unsigned short)(r >> 16);
  __bf16 o;
  __builtin_memcpy(&o, &h, sizeof(o));
  return o;
}

// ---------------------------------------------------------------------------
// K1: 8x8 max-pool of mask [16,1,256,256] -> p [16,1024]
// ---------------------------------------------------------------------------
__global__ __launch_bounds__(256) void cam_maskpool(const float* __restrict__ mask,
                                                    float* __restrict__ pfb) {
  int idx = blockIdx.x * 256 + threadIdx.x;            // 16384 total
  int b = idx >> 10, patch = idx & 1023;
  int py = patch >> 5, px = patch & 31;
  const float* src = mask + ((size_t)b * 256 + py * 8) * 256 + px * 8;
  float m = -3.4e38f;
#pragma unroll
  for (int r = 0; r < 8; ++r) {
    float4 a = *reinterpret_cast<const float4*>(src + r * 256);
    float4 c = *reinterpret_cast<const float4*>(src + r * 256 + 4);
    m = fmaxf(m, fmaxf(fmaxf(a.x, a.y), fmaxf(a.z, a.w)));
    m = fmaxf(m, fmaxf(fmaxf(c.x, c.y), fmaxf(c.z, c.w)));
  }
  pfb[idx] = m;
}

// ---------------------------------------------------------------------------
// K2: 2x2 avg-pool of feature_attn [16,128,64,64] + L2-normalize each patch row
//     -> fn bf16 [16,1024,128]  (fn fn^T == cosine matrix)
// ---------------------------------------------------------------------------
__global__ __launch_bounds__(128) void cam_fanorm(const float* __restrict__ fattn,
                                                  __bf16* __restrict__ fnb) {
  __shared__ float red[128];
  const int patch = blockIdx.x, b = blockIdx.y;
  const int py = patch >> 5, px = patch & 31;
  const int c = threadIdx.x;
  const float* src = fattn + (((size_t)(b * 128 + c) * 64) + 2 * py) * 64 + 2 * px;
  float v = 0.25f * (src[0] + src[1] + src[64] + src[65]);
  red[c] = v * v;
  __syncthreads();
#pragma unroll
  for (int o = 64; o; o >>= 1) {
    if (c < o) red[c] += red[c + o];
    __syncthreads();
  }
  float inv = rsqrtf(red[0]);
  fnb[((size_t)(b * 1024 + patch)) * 128 + c] = f2bf(v * inv);
}

// ---------------------------------------------------------------------------
// K3: per 16-row tile: cos = fn fn^T via WMMA bf16 (K=128 -> 4 chained WMMAs),
//     cmat = cos * p_i * (1-p_j) -> LDS [16][1024] f32, row softmax (fp32),
//     s = softmax * p_i * (1-p_j) -> bf16 S [16,1024,1024]
// Grid: (64 row-tiles, 16 batches), block = 256 threads = 8 waves.
// ---------------------------------------------------------------------------
__global__ __launch_bounds__(256) void cam_attn_softmax(const __bf16* __restrict__ fnb,
                                                        const float* __restrict__ pfb,
                                                        __bf16* __restrict__ S) {
  __shared__ float cmat[16][1024];   // 64 KB (gfx1250: 320 KB LDS per WGP)
  __shared__ float wred[8];
  __shared__ float bcast;

  const int b  = blockIdx.y;
  const int i0 = blockIdx.x * 16;
  const int tid = threadIdx.x, lane = tid & 31, wid = tid >> 5;
  const int h = lane >> 4, ln = lane & 15;

  // A fragments: rows i0..i0+15, all K=128 (4 chunks of 32).
  // A 16x32 layout: lane holds M=lane&15; K = h*16 + elem (matches B)
  // -> per lane the 16 bf16 are one contiguous 16-element run at h*16.
  B16x16 a[4];
  const __bf16* fnrow = fnb + ((size_t)b * 1024 + i0 + ln) * 128;
#pragma unroll
  for (int kc = 0; kc < 4; ++kc) {
    a[kc].q[0] = *reinterpret_cast<const uint4*>(fnrow + kc * 32 + h * 16);
    a[kc].q[1] = *reinterpret_cast<const uint4*>(fnrow + kc * 32 + h * 16 + 8);
  }
  float pi[8];
#pragma unroll
  for (int v = 0; v < 8; ++v) pi[v] = pfb[b * 1024 + i0 + h * 8 + v];

  // Each wave covers 8 of the 64 column tiles.
  for (int jt = wid; jt < 64; jt += 8) {
    v8f acc = {};
    // B 32x16 layout: lane holds N=lane&15; K = h*16 + elem -> one contiguous
    // 16-element (32 B) run per lane per chunk.
    const __bf16* fncol = fnb + ((size_t)b * 1024 + jt * 16 + ln) * 128;
#pragma unroll
    for (int kc = 0; kc < 4; ++kc) {
      B16x16 bb;
      bb.q[0] = *reinterpret_cast<const uint4*>(fncol + kc * 32 + h * 16);
      bb.q[1] = *reinterpret_cast<const uint4*>(fncol + kc * 32 + h * 16 + 8);
      acc = __builtin_amdgcn_wmma_f32_16x16x32_bf16(false, a[kc].v, false, bb.v,
                                                    (short)0, acc, false, false);
    }
    float pj1m = 1.0f - pfb[b * 1024 + jt * 16 + ln];
#pragma unroll
    for (int v = 0; v < 8; ++v)
      cmat[h * 8 + v][jt * 16 + ln] = acc[v] * pi[v] * pj1m;   // C/D: M=v+h*8, N=ln
  }
  __syncthreads();

  // fp32 row softmax over 1024 cols, then * p_i*(1-p_j) -> bf16 S
  for (int r = 0; r < 16; ++r) {
    float m = -3.4e38f;
    for (int t = tid; t < 1024; t += 256) m = fmaxf(m, cmat[r][t]);
#pragma unroll
    for (int o = 16; o; o >>= 1) m = fmaxf(m, __shfl_xor(m, o, 32));
    if (lane == 0) wred[wid] = m;
    __syncthreads();
    if (tid == 0) {
      float mm = wred[0];
#pragma unroll
      for (int i = 1; i < 8; ++i) mm = fmaxf(mm, wred[i]);
      bcast = mm;
    }
    __syncthreads();
    const float M = bcast;
    float s = 0.0f;
    for (int t = tid; t < 1024; t += 256) s += __expf(cmat[r][t] - M);
#pragma unroll
    for (int o = 16; o; o >>= 1) s += __shfl_xor(s, o, 32);
    if (lane == 0) wred[wid] = s;
    __syncthreads();
    if (tid == 0) {
      float ss = 0.0f;
#pragma unroll
      for (int i = 0; i < 8; ++i) ss += wred[i];
      bcast = ss;
    }
    __syncthreads();
    const float inv = 1.0f / bcast;
    const float prow = pfb[b * 1024 + i0 + r];
    __bf16* Srow = S + ((size_t)(b * 1024 + i0 + r)) * 1024;
    for (int t = tid; t < 1024; t += 256) {
      float v = __expf(cmat[r][t] - M) * inv * prow * (1.0f - pfb[b * 1024 + t]);
      Srow[t] = f2bf(v);
    }
    __syncthreads();
  }
}

// ---------------------------------------------------------------------------
// K4: out = S @ fp, fp = feature rearranged [1024 x 4096]
// Column permutation: tile t = (ph, ch_blk); within tile n' = pw*8 + c'
//   d = ph*512 + pw*64 + ch_blk*8 + c'
// -> epilogue writes 256 contiguous floats per (ch,y) row (coalesced stores).
// Block = 256 thr (8 waves) computes a 64(i) x 64(n') tile; K loop of 32 with
// DOUBLE-BUFFERED LDS: S tile staged via async global->LDS b128 (ASYNCcnt),
// feature tile loaded+bf16-converted in VALU; both overlap current WMMAs.
// Grid: (64 col-tiles, 16 row-tiles, 16 batches)
// ---------------------------------------------------------------------------
__global__ __launch_bounds__(256) void cam_attn_transfer(const float* __restrict__ feature,
                                                         const __bf16* __restrict__ S,
                                                         float* __restrict__ out) {
  __shared__ __bf16 lA[2][64][32];   // S tile  [i][k], double-buffered
  __shared__ __bf16 lBt[2][64][32];  // B tile transposed [n'][k], double-buffered
  __shared__ float  lC[64][64];      // f32 epilogue staging

  const int b  = blockIdx.z;
  const int i0 = blockIdx.y * 64;
  const int t  = blockIdx.x;                  // 0..63
  const int ph = t >> 3, ch0 = (t & 7) * 8;
  const int tid = threadIdx.x, lane = tid & 31, wid = tid >> 5;
  const int h = lane >> 4, ln = lane & 15;
  const int isub = wid >> 1, dpair = wid & 1;
  const __bf16* Sb = S + (size_t)b * 1024 * 1024;

  v8f acc0 = {}, acc1 = {};

  // Per-thread staging coordinates (constant over the K loop).
  const int sr = tid >> 2, scs = (tid & 3) * 8;          // S tile row/col-seg
  const int ci = tid >> 5;                               // feature channel 0..7
  const int xl = (tid & 31) * 8;                         // feature x base
  const int kcol = tid & 31;                             // = px
  const float* fbase =
      feature + (((size_t)(b * 64 + ch0 + ci) * 256)) * 256 + xl;

  auto stage = [&](int buf, int kk) {
    // --- S tile: 64x32 bf16, one b128 per thread, async global->LDS ---
    const __bf16* gsrc = Sb + (size_t)(i0 + sr) * 1024 + kk * 32 + scs;
#if HAVE_ASYNC_LDS
    __builtin_amdgcn_global_load_async_to_lds_b128(
        (gv4i_p)gsrc, (lv4i_p)&lA[buf][sr][scs], 0, 0);
#else
    *reinterpret_cast<uint4*>(&lA[buf][sr][scs]) =
        *reinterpret_cast<const uint4*>(gsrc);
#endif
    // --- feature tile: 8ch x 256x (y = kk*8+ph), f32 -> bf16, transposed ---
    const float* src = fbase + (size_t)(kk * 8 + ph) * 256;
    float4 f0 = *reinterpret_cast<const float4*>(src);
    float4 f1 = *reinterpret_cast<const float4*>(src + 4);
    // Prefetch the feature row two k-steps ahead (global_prefetch_b8).
    int kpf = (kk + 2 < 32) ? (kk + 2) : 31;
    __builtin_prefetch(fbase + (size_t)(kpf * 8 + ph) * 256, 0, 1);
    lBt[buf][0 * 8 + ci][kcol] = f2bf(f0.x);
    lBt[buf][1 * 8 + ci][kcol] = f2bf(f0.y);
    lBt[buf][2 * 8 + ci][kcol] = f2bf(f0.z);
    lBt[buf][3 * 8 + ci][kcol] = f2bf(f0.w);
    lBt[buf][4 * 8 + ci][kcol] = f2bf(f1.x);
    lBt[buf][5 * 8 + ci][kcol] = f2bf(f1.y);
    lBt[buf][6 * 8 + ci][kcol] = f2bf(f1.z);
    lBt[buf][7 * 8 + ci][kcol] = f2bf(f1.w);
  };

  // Prologue: fill buffer 0 for kk=0.
  stage(0, 0);
  wait_async0();
  __syncthreads();

  for (int kk = 0; kk < 32; ++kk) {          // j = kk*32 + k, so py=kk, px=k
    const int cur = kk & 1;
    if (kk + 1 < 32) stage(cur ^ 1, kk + 1); // overlap next copy with WMMAs

    // A fragment (two 16B ds loads per lane)
    B16x16 af;
    {
      const __bf16* arow = &lA[cur][isub * 16 + ln][0];
      af.q[0] = *reinterpret_cast<const uint4*>(arow + h * 16);
      af.q[1] = *reinterpret_cast<const uint4*>(arow + h * 16 + 8);
    }
    // B fragments (contiguous 32B per lane thanks to [n'][k] layout)
    B16x16 b0, b1;
    {
      const __bf16* br0 = &lBt[cur][dpair * 32 + ln][0];
      const __bf16* br1 = &lBt[cur][dpair * 32 + 16 + ln][0];
      b0.q[0] = *reinterpret_cast<const uint4*>(br0 + h * 16);
      b0.q[1] = *reinterpret_cast<const uint4*>(br0 + h * 16 + 8);
      b1.q[0] = *reinterpret_cast<const uint4*>(br1 + h * 16);
      b1.q[1] = *reinterpret_cast<const uint4*>(br1 + h * 16 + 8);
    }
    acc0 = __builtin_amdgcn_wmma_f32_16x16x32_bf16(false, af.v, false, b0.v,
                                                   (short)0, acc0, false, false);
    acc1 = __builtin_amdgcn_wmma_f32_16x16x32_bf16(false, af.v, false, b1.v,
                                                   (short)0, acc1, false, false);

    wait_async0();          // next buffer's async copy complete (ASYNCcnt)
    __syncthreads();        // + all waves' ds stores visible (s_wait_dscnt)
  }

  // Epilogue: stage in LDS then fully coalesced 256-float output rows.
#pragma unroll
  for (int v = 0; v < 8; ++v) {
    int row = isub * 16 + h * 8 + v;
    lC[row][dpair * 32 + ln]      = acc0[v];
    lC[row][dpair * 32 + 16 + ln] = acc1[v];
  }
  __syncthreads();

#pragma unroll
  for (int rr = 0; rr < 16; ++rr) {
    int py_rel = rr >> 3, cc = rr & 7;
    int y = ((i0 >> 5) + py_rel) * 8 + ph;
    int x = tid;                              // coalesced: lane == x
    int px = x >> 3, pw = x & 7;
    out[(((size_t)(b * 64 + ch0 + cc) * 256) + y) * 256 + x] =
        lC[py_rel * 32 + px][pw * 8 + cc];
  }
}

// ---------------------------------------------------------------------------
extern "C" void kernel_launch(void* const* d_in, const int* in_sizes, int n_in,
                              void* d_out, int out_size, void* d_ws, size_t ws_size,
                              hipStream_t stream) {
  const float* feature = (const float*)d_in[0];   // [16,64,256,256]
  const float* fattn   = (const float*)d_in[1];   // [16,128,64,64]
  const float* mask    = (const float*)d_in[2];   // [16,1,256,256]
  float* out = (float*)d_out;                     // [16,64,256,256]

  char* ws = (char*)d_ws;
  float*  pfb = (float*)ws;                                   // 64 KB
  __bf16* fnb = (__bf16*)(ws + 65536);                        // 4 MB
  __bf16* S   = (__bf16*)(ws + 65536 + (size_t)16 * 1024 * 128 * 2);  // 32 MB

  cam_maskpool<<<64, 256, 0, stream>>>(mask, pfb);
  cam_fanorm<<<dim3(1024, 16), 128, 0, stream>>>(fattn, fnb);
  cam_attn_softmax<<<dim3(64, 16), 256, 0, stream>>>(fnb, pfb, S);
  cam_attn_transfer<<<dim3(64, 16, 16), 256, 0, stream>>>(feature, S, out);
}
